// FastPaperMambaBlock_8263517077567
// MI455X (gfx1250) — matrix operand order repaired and checked by reading, compiled, not run on hardware
//
#include <hip/hip_runtime.h>
#include <math.h>

// ---------------------------------------------------------------------------
// CDNA5 WMMA types (wave32). v_wmma_f32_16x16x32_bf16:
//   A: 16x32 bf16, B: 32x16 bf16 (v16bf per lane), C/D: 16x16 f32 (v8f)
// ---------------------------------------------------------------------------
typedef __attribute__((ext_vector_type(16))) __bf16    v16bf;
typedef __attribute__((ext_vector_type(2)))  __bf16    v2bf;
typedef __attribute__((ext_vector_type(8)))  float     v8f;
typedef __attribute__((ext_vector_type(8)))  unsigned  v8u;
typedef __attribute__((ext_vector_type(4)))  int       v4i;

union FragBF { v8u u; v16bf h; };

static __device__ __forceinline__ unsigned f32u(float f) {
    union { float f; unsigned u; } v; v.f = f; return v.u;
}

// pack two f32 -> packed bf16 (hi|lo). Preference:
//   1) v_cvt_pk_bf16_f32 (1 instr)   2) add+add+v_perm_b32 (3 instr, round-half-up)
//   3) shift/mask fallback
#if __has_builtin(__builtin_amdgcn_cvt_pk_bf16_f32)
static __device__ __forceinline__ unsigned pack2bf(float lo, float hi) {
    union { v2bf h; unsigned u; } c;
    c.h = __builtin_amdgcn_cvt_pk_bf16_f32(lo, hi);
    return c.u;
}
#elif __has_builtin(__builtin_amdgcn_perm)
static __device__ __forceinline__ unsigned pack2bf(float lo, float hi) {
    // take bytes [3:2] of each rounded value: dst = {hi.b3,hi.b2,lo.b3,lo.b2}
    return __builtin_amdgcn_perm(f32u(hi) + 0x8000u, f32u(lo) + 0x8000u,
                                 0x07060302u);
}
#else
static __device__ __forceinline__ unsigned pack2bf(float lo, float hi) {
    return ((f32u(lo) + 0x8000u) >> 16) | ((f32u(hi) + 0x8000u) & 0xFFFF0000u);
}
#endif

#if __has_builtin(__builtin_amdgcn_global_load_async_to_lds_b128) && \
    __has_builtin(__builtin_amdgcn_s_wait_asynccnt)
#define USE_ASYNC_LDS 1
#else
#define USE_ASYNC_LDS 0
#endif

// exact parameter types per compiler diagnostic:
//   (v4i addrspace(1)* src, v4i addrspace(3)* dst, imm offset, imm cpol)
#define AS_GLB4(p) ((__attribute__((address_space(1))) v4i*)(p))
#define AS_LDS4(p) ((__attribute__((address_space(3))) v4i*)(p))

static __device__ __forceinline__ float silu_f(float v) {
    return v / (1.0f + __expf(-v));
}

// ---------------------------------------------------------------------------
// C = A(MxK) * W^T (W: NxK row-major) + bias.  fp32 in/out, bf16 WMMA compute.
// REQUIRES: M % 64 == 0, N % 64 == 0, K % 32 == 0 (no edge guards).
// Block: 128 threads (4 waves), 64x64 tile; wave = 32x32 = 2x2 WMMA tiles.
// MODE 0: +bias     MODE 2: softplus+clip(0.001,0.2)    MODE 3: +bias+resid
// ---------------------------------------------------------------------------
template <int MODE>
__global__ __launch_bounds__(128)
void wmma_gemm_bf16(const float* __restrict__ A, const float* __restrict__ W,
                    const float* __restrict__ bias, const float* __restrict__ resid,
                    float* __restrict__ C, int M, int N, int K, int ldc)
{
    // fp32 staging tiles (pure copies -> async-LDS friendly); 16 KB total
    __shared__ float As[64][32];        // [m][k]
    __shared__ float Bs[64][32];        // [n][k]

    const int tid  = threadIdx.x;
    const int lane = tid & 31;
    const int w    = tid >> 5;
    const int wm   = w >> 1;
    const int wn   = w & 1;
    const int row0 = blockIdx.y * 64;
    const int col0 = blockIdx.x * 64;

    const int half = lane >> 4;         // 0: lanes 0-15, 1: lanes 16-31
    const int l16  = lane & 15;

    // staging assignment: thread -> (row lr, 16-float k-chunk lk)
    const int lr = tid >> 1;            // 0..63
    const int lk = (tid & 1) * 16;      // 0 or 16 (floats)

    const float* __restrict__ Arow = A + (size_t)(row0 + lr) * K + lk;
    const float* __restrict__ Wrow = W + (size_t)(col0 + lr) * K + lk;

    v8f acc[2][2] = {};

    for (int k0 = 0; k0 < K; k0 += 32) {
        __syncthreads();
#if USE_ASYNC_LDS
        // pure DMA-style copy: 4x16B per tile per thread, no VGPR round-trip
        #pragma unroll
        for (int i = 0; i < 4; ++i) {
            __builtin_amdgcn_global_load_async_to_lds_b128(
                AS_GLB4(Arow + k0 + 4 * i), AS_LDS4(&As[lr][lk + 4 * i]), 0, 0);
            __builtin_amdgcn_global_load_async_to_lds_b128(
                AS_GLB4(Wrow + k0 + 4 * i), AS_LDS4(&Bs[lr][lk + 4 * i]), 0, 0);
        }
        __builtin_amdgcn_s_wait_asynccnt(0);
#else
        {
            const float4* ap = (const float4*)(Arow + k0);
            const float4* bp = (const float4*)(Wrow + k0);
            #pragma unroll
            for (int i = 0; i < 4; ++i) {
                *(float4*)&As[lr][lk + 4 * i] = ap[i];
                *(float4*)&Bs[lr][lk + 4 * i] = bp[i];
            }
        }
#endif
        __syncthreads();

        // B fragments: column n = l16; lanes 0-15 -> K 0..15, 16-31 -> 16..31.
        FragBF bf[2];
        #pragma unroll
        for (int nt = 0; nt < 2; ++nt) {
            const int n    = wn * 32 + nt * 16 + l16;
            const int koff = 16 * half;
            #pragma unroll
            for (int j = 0; j < 8; ++j)
                bf[nt].u[j] = pack2bf(Bs[n][koff + 2 * j], Bs[n][koff + 2 * j + 1]);
        }

        #pragma unroll
        for (int mt = 0; mt < 2; ++mt) {
            // A fragment: row m = l16; K pairs: 2j+16h
            const int m = wm * 32 + mt * 16 + l16;
            FragBF af;
            #pragma unroll
            for (int j = 0; j < 8; ++j) {
                const int kb = 2 * j + 16 * half;
                af.u[j] = pack2bf(As[m][kb], As[m][kb + 1]);
            }
            #pragma unroll
            for (int nt = 0; nt < 2; ++nt) {
                acc[mt][nt] = __builtin_amdgcn_wmma_f32_16x16x32_bf16(
                    false, af.h, false, bf[nt].h,
                    (short)0, acc[mt][nt], false, false);
            }
        }
    }

    // Epilogue. C/D layout: VGPR j -> M = j + 8*half, N = l16.
    #pragma unroll
    for (int nt = 0; nt < 2; ++nt) {
        const int n    = col0 + wn * 32 + nt * 16 + l16;
        const float bv = bias[n];
        #pragma unroll
        for (int mt = 0; mt < 2; ++mt) {
            #pragma unroll
            for (int j = 0; j < 8; ++j) {
                const int m = row0 + wm * 32 + mt * 16 + half * 8 + j;
                float v = acc[mt][nt][j] + bv;
                if constexpr (MODE == 2) {          // softplus then clip
                    v = (v > 20.0f) ? v : log1pf(__expf(v));
                    v = fminf(fmaxf(v, 0.001f), 0.2f);
                } else if constexpr (MODE == 3) {   // + residual
                    v += resid[(size_t)m * ldc + n];
                }
                C[(size_t)m * ldc + n] = v;
            }
        }
    }
}

// ---------------------------------------------------------------------------
// Pack W_bc = [B_w ; C_w] (64 x K) and bias_bc = [B_b ; C_b] (64)
// ---------------------------------------------------------------------------
__global__ __launch_bounds__(256)
void pack_bc_kernel(const float* __restrict__ Bw, const float* __restrict__ Cw,
                    const float* __restrict__ Bb, const float* __restrict__ Cb,
                    float* __restrict__ Wbc, float* __restrict__ bbc, int N, int K)
{
    int idx = blockIdx.x * blockDim.x + threadIdx.x;
    int total = 2 * N * K;
    if (idx < total) {
        int n = idx / K, k = idx % K;
        Wbc[idx] = (n < N) ? Bw[n * K + k] : Cw[(n - N) * K + k];
    }
    if (idx < 2 * N) bbc[idx] = (idx < N) ? Bb[idx] : Cb[idx - N];
}

// ---------------------------------------------------------------------------
// Depthwise conv1d (K=4, taps t-2..t+1) + bias + SiLU.
// ---------------------------------------------------------------------------
__global__ __launch_bounds__(256)
void conv_silu_kernel(const float* __restrict__ xz,
                      const float* __restrict__ conv_w,
                      const float* __restrict__ conv_b,
                      float* __restrict__ xs, int Bn, int L, int D)
{
    int idx = blockIdx.x * blockDim.x + threadIdx.x;
    int total = Bn * L * D;
    if (idx >= total) return;
    int d = idx % D;
    int t = (idx / D) % L;
    int b = idx / (D * L);
    float acc = conv_b[d];
    #pragma unroll
    for (int k = 0; k < 4; ++k) {
        int tt = t + k - 2;
        if (tt >= 0 && tt < L)
            acc += conv_w[d * 4 + k] * xz[(size_t)(b * L + tt) * (2 * D) + d];
    }
    xs[idx] = silu_f(acc);
}

// ---------------------------------------------------------------------------
// Selective scan. One wave32 per (b, d); lane = n (N == 32).
// BC holds [Bm | Cm] rows of width 64.
// ---------------------------------------------------------------------------
__global__ __launch_bounds__(128)
void scan_kernel(const float* __restrict__ xs, const float* __restrict__ dt,
                 const float* __restrict__ BC, const float* __restrict__ A_log,
                 float* __restrict__ y, int Bn, int L, int D, int N)
{
    int wid  = blockIdx.x * (blockDim.x >> 5) + (threadIdx.x >> 5);
    int lane = threadIdx.x & 31;
    int d = wid % D;
    int b = wid / D;
    if (b >= Bn) return;

    float al = A_log[d * N + lane];
    al = fminf(fmaxf(al, -5.0f), 2.0f);
    const float Aval = -__expf(al);

    float state = 0.0f;
    const size_t base = (size_t)b * L;
    for (int t = 0; t < L; ++t) {
        const size_t rt = base + t;
        float xv  = xs[rt * D + d];
        float dtv = dt[rt * D + d];
        float Bv  = BC[rt * 64 + lane];
        float Cv  = BC[rt * 64 + N + lane];
        float abar = __expf(dtv * Aval);
        state = state * abar + dtv * Bv * xv;
        state = fminf(fmaxf(state, -10.0f), 10.0f);
        float yv = Cv * state;
        #pragma unroll
        for (int off = 16; off > 0; off >>= 1)
            yv += __shfl_xor(yv, off, 32);
        if (lane == 0) y[rt * D + d] = yv;
    }
}

// ---------------------------------------------------------------------------
// Gating: yg = y * silu(z)
// ---------------------------------------------------------------------------
__global__ __launch_bounds__(256)
void gate_kernel(const float* __restrict__ y, const float* __restrict__ xz,
                 float* __restrict__ yg, int total, int D)
{
    int idx = blockIdx.x * blockDim.x + threadIdx.x;
    if (idx >= total) return;
    int d = idx % D;
    int m = idx / D;
    float z = xz[(size_t)m * (2 * D) + D + d];
    yg[idx] = y[idx] * silu_f(z);
}

// ---------------------------------------------------------------------------
// LayerNorm over D per row; one block per row.
// ---------------------------------------------------------------------------
__global__ __launch_bounds__(256)
void ln_kernel(const float* __restrict__ pre, const float* __restrict__ g,
               const float* __restrict__ beta, float* __restrict__ out, int D)
{
    __shared__ float red[256];
    const int m = blockIdx.x;
    const float* row = pre + (size_t)m * D;

    float s = 0.0f, s2 = 0.0f;
    for (int i = threadIdx.x; i < D; i += 256) {
        float v = row[i];
        s += v; s2 += v * v;
    }
    red[threadIdx.x] = s;
    __syncthreads();
    for (int o = 128; o > 0; o >>= 1) {
        if (threadIdx.x < o) red[threadIdx.x] += red[threadIdx.x + o];
        __syncthreads();
    }
    float mean = red[0] / (float)D;
    __syncthreads();
    red[threadIdx.x] = s2;
    __syncthreads();
    for (int o = 128; o > 0; o >>= 1) {
        if (threadIdx.x < o) red[threadIdx.x] += red[threadIdx.x + o];
        __syncthreads();
    }
    float var = red[0] / (float)D - mean * mean;
    float rstd = rsqrtf(var + 1e-5f);
    for (int i = threadIdx.x; i < D; i += 256)
        out[(size_t)m * D + i] = g[i] * (row[i] - mean) * rstd + beta[i];
}

// ---------------------------------------------------------------------------
// Host launcher
// ---------------------------------------------------------------------------
extern "C" void kernel_launch(void* const* d_in, const int* in_sizes, int n_in,
                              void* d_out, int out_size, void* d_ws, size_t ws_size,
                              hipStream_t stream)
{
    const float* x         = (const float*)d_in[0];
    const float* in_proj_w = (const float*)d_in[1];
    const float* in_proj_b = (const float*)d_in[2];
    const float* conv_w    = (const float*)d_in[3];
    const float* conv_b    = (const float*)d_in[4];
    const float* A_log     = (const float*)d_in[5];
    const float* B_w       = (const float*)d_in[6];
    const float* B_b       = (const float*)d_in[7];
    const float* C_w       = (const float*)d_in[8];
    const float* C_b       = (const float*)d_in[9];
    const float* dt_w      = (const float*)d_in[10];
    const float* dt_b      = (const float*)d_in[11];
    const float* out_w     = (const float*)d_in[12];
    const float* out_b     = (const float*)d_in[13];
    const float* ln_g      = (const float*)d_in[14];
    const float* ln_beta   = (const float*)d_in[15];

    constexpr int Bn = 2, L = 2048, D = 1024, N = 32;
    constexpr int M  = Bn * L;                          // 4096 rows

    // Workspace carve-up (~113 MB total)
    char*  ws   = (char*)d_ws;
    float* xz   = (float*)ws; ws += (size_t)M * 2 * D * sizeof(float); // 32 MB
    float* xs   = (float*)ws; ws += (size_t)M * D * sizeof(float);     // 16 MB
    float* BC   = (float*)ws; ws += (size_t)M * 64 * sizeof(float);    // 1 MB
    float* dtb  = (float*)ws; ws += (size_t)M * D * sizeof(float);     // 16 MB
    float* y    = (float*)ws; ws += (size_t)M * D * sizeof(float);     // 16 MB
    float* yg   = (float*)ws; ws += (size_t)M * D * sizeof(float);     // 16 MB
    float* pre  = (float*)ws; ws += (size_t)M * D * sizeof(float);     // 16 MB
    float* Wbc  = (float*)ws; ws += (size_t)64 * D * sizeof(float);    // 256 KB
    float* bbc  = (float*)ws; ws += 256;                               // 64 floats

    dim3 blk(128);

    // 0) pack fused [B_w;C_w] weights
    pack_bc_kernel<<<(2 * N * D + 255) / 256, 256, 0, stream>>>(
        B_w, C_w, B_b, C_b, Wbc, bbc, N, D);

    // 1) xz = x @ in_proj_w^T + in_proj_b            (4096 x 2048, K=1024)
    wmma_gemm_bf16<0><<<dim3((2 * D) / 64, M / 64), blk, 0, stream>>>(
        x, in_proj_w, in_proj_b, nullptr, xz, M, 2 * D, D, 2 * D);

    // 2) xs = silu(depthwise_conv(xc) + conv_b)
    conv_silu_kernel<<<(M * D + 255) / 256, 256, 0, stream>>>(
        xz, conv_w, conv_b, xs, Bn, L, D);

    // 3) BC = xs @ [B_w;C_w]^T + [B_b;C_b]           (4096 x 64)
    wmma_gemm_bf16<0><<<dim3(1, M / 64), blk, 0, stream>>>(
        xs, Wbc, bbc, nullptr, BC, M, 64, D, 64);

    // 4) dt = clip(softplus(xs @ dt_w^T + dt_b))     (4096 x 1024)
    wmma_gemm_bf16<2><<<dim3(D / 64, M / 64), blk, 0, stream>>>(
        xs, dt_w, dt_b, nullptr, dtb, M, D, D, D);

    // 5) selective scan: one wave per (b,d), lanes = n
    scan_kernel<<<(Bn * D) / 4, 128, 0, stream>>>(
        xs, dtb, BC, A_log, y, Bn, L, D, N);

    // 6) yg = y * silu(z)
    gate_kernel<<<(M * D + 255) / 256, 256, 0, stream>>>(
        y, xz, yg, M * D, D);

    // 7) pre = yg @ out_w^T + out_b + residual(x)    (4096 x 1024)
    wmma_gemm_bf16<3><<<dim3(D / 64, M / 64), blk, 0, stream>>>(
        yg, out_w, out_b, x, pre, M, D, D, D);

    // 8) LayerNorm -> d_out
    ln_kernel<<<M, 256, 0, stream>>>(pre, ln_g, ln_beta, (float*)d_out, D);
}
